// DeepMemoryLevel_34754875359826
// MI455X (gfx1250) — matrix operand order repaired and checked by reading, not compiled, not run on hardware
//
#include <hip/hip_runtime.h>
#include <hip/hip_bf16.h>
#include <math.h>

typedef _Float16 half_t;
typedef __attribute__((ext_vector_type(16))) _Float16 v16h;
typedef __attribute__((ext_vector_type(8)))  _Float16 v8h;
typedef __attribute__((ext_vector_type(8)))  float    v8f;

static constexpr int TT    = 2048;
static constexpr int DIM   = 1024;
static constexpr int MEM   = 256;
static constexpr int POLY  = 512;
static constexpr int HID   = 1024;
static constexpr int CHUNK = 32;
static constexpr int NC    = TT / CHUNK;
static constexpr int NPER  = 4;
static constexpr int SEQ   = 2044;
static constexpr int KCONV = 4;
static constexpr float MAXGN = 10.0f;

// ---------------- device helpers ----------------

__device__ __forceinline__ float sigmoidf_(float x) { return 1.0f / (1.0f + __expf(-x)); }
__device__ __forceinline__ float siluf_(float x)    { return x * sigmoidf_(x); }

__device__ __forceinline__ v8f wmma32(v16h a, v16h b, v8f c) {
  return __builtin_amdgcn_wmma_f32_16x16x32_f16(false, a, false, b, (short)0, c, false, false);
}

__device__ __forceinline__ v16h cat16(v8h lo, v8h hi) {
  return __builtin_shufflevector(lo, hi, 0, 1, 2, 3, 4, 5, 6, 7,
                                 8, 9, 10, 11, 12, 13, 14, 15);
}

// A fragment 16x32 (MxK) from f16 row-major src (pre-offset to tile origin), ld in halfs.
// Per lane: two contiguous 8-half runs -> 2x global_load_b128.
// element i -> K = i + ((lane>>4)<<4)
__device__ __forceinline__ v16h load_a16v(const half_t* __restrict__ src, int ld) {
  int lane = threadIdx.x & 31;
  int m  = lane & 15;
  int kb = (lane >> 4) << 4;
  const half_t* p = src + (size_t)m * ld + kb;
  v8h lo = *(const v8h*)(p);        // K = kb .. kb+7
  v8h hi = *(const v8h*)(p + 8);    // K = kb+8 .. kb+15
  return cat16(lo, hi);
}

// B fragment 32x16 (KxN) from *N-major* f16 storage BT[n*ldk + k] (pre-offset to k origin).
// Per lane: 16 contiguous halfs -> 2x global_load_b128.
// element i -> K = i + ((lane>=16)<<4), N = lane&15
__device__ __forceinline__ v16h load_b16t(const half_t* __restrict__ bt, int ldk) {
  int lane = threadIdx.x & 31;
  int n  = lane & 15;
  int kb = (lane >> 4) << 4;
  const half_t* p = bt + (size_t)n * ldk + kb;
  v8h lo = *(const v8h*)(p);
  v8h hi = *(const v8h*)(p + 8);
  return cat16(lo, hi);
}

// Scaled B fragment from N-major f32 storage srcT[n*CHUNK + k] (k = token 0..31),
// folding diag(scale[k]) into the load. Two v8f (b128) loads per lane.
__device__ __forceinline__ v16h load_b16sT(const float* __restrict__ srcT, int n0,
                                           const float* scale) {
  int lane = threadIdx.x & 31;
  int col = n0 + (lane & 15);
  int kb  = (lane >> 4) << 4;
  const float* p = srcT + (size_t)col * CHUNK + kb;
  v8f s0 = *(const v8f*)(p);
  v8f s1 = *(const v8f*)(p + 8);
  v16h b;
#pragma unroll
  for (int i = 0; i < 8; ++i) {
    b[i]     = (half_t)(s0[i] * scale[kb + i]);
    b[i + 8] = (half_t)(s1[i] * scale[kb + 8 + i]);
  }
  return b;
}

// ---------------- generic WMMA GEMM: C(f32,MxN) = A(f16,MxK) @ B, B given N-major ----
// one 16x64 strip per wave (1x4 register blocking); waves == (M/16)*(N/64)
__global__ __launch_bounds__(256) void gemm_f16_kernel(const half_t* __restrict__ A,
                                                       const half_t* __restrict__ BT,
                                                       float* __restrict__ C,
                                                       int M, int N, int K) {
  int gw  = (int)((blockIdx.x * blockDim.x + threadIdx.x) >> 5);
  int ncg = N >> 6;
  int tm  = (gw / ncg) << 4;
  int tn0 = (gw % ncg) << 6;
  if (tm >= M) return;
  v8f acc0 = {}, acc1 = {}, acc2 = {}, acc3 = {};
  const half_t* ap = A + (size_t)tm * K;
  for (int k = 0; k < K; k += 32) {
    __builtin_prefetch(ap + k + 256, 0, 3);
    v16h a = load_a16v(ap + k, K);
    acc0 = wmma32(a, load_b16t(BT + (size_t)(tn0 +  0) * K + k, K), acc0);
    acc1 = wmma32(a, load_b16t(BT + (size_t)(tn0 + 16) * K + k, K), acc1);
    acc2 = wmma32(a, load_b16t(BT + (size_t)(tn0 + 32) * K + k, K), acc2);
    acc3 = wmma32(a, load_b16t(BT + (size_t)(tn0 + 48) * K + k, K), acc3);
  }
  int lane = threadIdx.x & 31;
  int n = lane & 15, mb = (lane >> 4) << 3;
#pragma unroll
  for (int i = 0; i < 8; ++i) {
    size_t row = (size_t)(tm + mb + i) * N + tn0 + n;
    C[row +  0] = acc0[i];
    C[row + 16] = acc1[i];
    C[row + 32] = acc2[i];
    C[row + 48] = acc3[i];
  }
}

// ---------------- small elementwise / setup kernels ----------------

__global__ void build_xf_kernel(const float* __restrict__ x, const float* __restrict__ pers,
                                float* __restrict__ xf32, half_t* __restrict__ xf16) {
  int i = blockIdx.x * 256 + threadIdx.x;
  if (i >= TT * DIM) return;
  int t = i / DIM, d = i - t * DIM;
  float v = (t < NPER) ? pers[t * DIM + d] : x[(size_t)(t - NPER) * DIM + d];
  xf32[i] = v;
  xf16[i] = (half_t)v;
}

// transpose-convert: in (RxC f32, row-major) -> out (CxR f16) so out is N-major B
__global__ void f32_to_f16_T_kernel(const float* __restrict__ in, half_t* __restrict__ out,
                                    int R, int C) {
  int i = blockIdx.x * 256 + threadIdx.x;
  if (i >= R * C) return;
  int r = i / C, c = i - r * C;
  out[(size_t)c * R + r] = (half_t)in[i];
}

__global__ void init_mem_kernel(const float* __restrict__ MW1, const float* __restrict__ MW2,
                                float* W1, float* W2, float* S1, float* S2,
                                half_t* W1Tm, half_t* W2h, half_t* W2Tm) {
  int i = blockIdx.x * 256 + threadIdx.x;
  if (i >= POLY * HID) return;
  float w1 = MW1[i];                     // MW1: POLY x HID
  W1[i] = w1; S1[i] = 0.0f;
  int p1 = i / HID, h1 = i - p1 * HID;
  W1Tm[(size_t)h1 * POLY + p1] = (half_t)w1;
  float w2 = MW2[i];                     // MW2: HID x POLY (same element count)
  W2[i] = w2; S2[i] = 0.0f;
  W2h[i] = (half_t)w2;
  int h2 = i / POLY, p2 = i - h2 * POLY;
  W2Tm[(size_t)p2 * HID + h2] = (half_t)w2;
}

__global__ void conv_poly_kernel(const float* __restrict__ kpre, const float* __restrict__ qpre,
                                 const float* __restrict__ vpre,
                                 const float* __restrict__ ckw, const float* __restrict__ cqw,
                                 const float* __restrict__ cvw,
                                 const float* __restrict__ ckb, const float* __restrict__ cqb,
                                 const float* __restrict__ cvb,
                                 const float* __restrict__ pc,
                                 half_t* __restrict__ phiK, half_t* __restrict__ phiKT,
                                 half_t* __restrict__ phiQ, half_t* __restrict__ v16o) {
  int i = blockIdx.x * 256 + threadIdx.x;
  if (i >= TT * MEM) return;
  int t = i / MEM, c = i - t * MEM;
  float c0 = pc[0], c1 = pc[1];
  float ak = ckb[c], aq = cqb[c], av = cvb[c];
#pragma unroll
  for (int j = 0; j < KCONV; ++j) {
    int ts = t - (KCONV - 1) + j;
    if (ts >= 0) {
      size_t s = (size_t)ts * MEM + c;
      ak += ckw[c * KCONV + j] * kpre[s];
      aq += cqw[c * KCONV + j] * qpre[s];
      av += cvw[c * KCONV + j] * vpre[s];
    }
  }
  float sk = siluf_(ak), sq = siluf_(aq), sv = siluf_(av);
  half_t k0 = (half_t)(c0 * sk), k1 = (half_t)(c1 * sk * sk);
  phiK[(size_t)t * POLY + c]        = k0;
  phiK[(size_t)t * POLY + MEM + c]  = k1;
  phiKT[(size_t)c * TT + t]         = k0;   // transposed copy for update-GEMM A^T loads
  phiKT[(size_t)(MEM + c) * TT + t] = k1;
  phiQ[(size_t)t * POLY + c]        = (half_t)(c0 * sq);
  phiQ[(size_t)t * POLY + MEM + c]  = (half_t)(c1 * sq * sq);
  v16o[(size_t)t * MEM + c]         = (half_t)sv;
}

__global__ __launch_bounds__(256) void kn2_kernel(const half_t* __restrict__ phiK,
                                                  float* __restrict__ kn2) {
  __shared__ float red[256];
  int t = blockIdx.x, tid = threadIdx.x;
  float s = 0.0f;
  for (int j = tid; j < POLY; j += 256) {
    float v = (float)phiK[(size_t)t * POLY + j];
    s += v * v;
  }
  red[tid] = s; __syncthreads();
  for (int st = 128; st > 0; st >>= 1) {
    if (tid < st) red[tid] += red[tid + st];
    __syncthreads();
  }
  if (tid == 0) kn2[t] = red[0];
}

__global__ __launch_bounds__(256) void gates_kernel(const float* __restrict__ xf,
    const float* lrw, const float* lrb, const float* momw, const float* momb,
    const float* decw, const float* decb, const float* gw, const float* gb,
    float* lr, float* eta, float* alpha, float* gate) {
  __shared__ float red[4][256];
  int t = blockIdx.x, tid = threadIdx.x;
  float a0 = 0, a1 = 0, a2 = 0, a3 = 0;
  const float* xr = xf + (size_t)t * DIM;
  for (int j = tid; j < DIM; j += 256) {
    float xv = xr[j];
    a0 += xv * lrw[j]; a1 += xv * momw[j]; a2 += xv * decw[j]; a3 += xv * gw[j];
  }
  red[0][tid] = a0; red[1][tid] = a1; red[2][tid] = a2; red[3][tid] = a3;
  __syncthreads();
  for (int st = 128; st > 0; st >>= 1) {
    if (tid < st)
      for (int r = 0; r < 4; ++r) red[r][tid] += red[r][tid + st];
    __syncthreads();
  }
  if (tid == 0) {
    lr[t]    = sigmoidf_(red[0][0] + lrb[0]);
    eta[t]   = sigmoidf_(red[1][0] + momb[0]);
    alpha[t] = sigmoidf_(red[2][0] + decb[0]);
    gate[t]  = sigmoidf_(red[3][0] + gb[0]);
  }
}

__global__ void gatefold_kernel(const float* __restrict__ retr, const float* __restrict__ gate,
                                half_t* __restrict__ retr16) {
  int i = blockIdx.x * 256 + threadIdx.x;
  if (i < TT * MEM) retr16[i] = (half_t)(retr[i] * gate[i / MEM]);
}

__global__ void copyout_kernel(const float* __restrict__ ybuf, float* __restrict__ out) {
  int i = blockIdx.x * 256 + threadIdx.x;
  if (i < SEQ * DIM) out[i] = ybuf[(size_t)NPER * DIM + i];
}

// ---------------- sequential chunk-scan kernel: 1 block, 512 threads (16 waves) ----------------

__device__ __forceinline__ void rownorm2_512(const float* __restrict__ buf, int rowlen,
                                             float* shpart, float* out32) {
  int tid = threadIdx.x;
  int t = tid & 31, seg = tid >> 5;        // 16 segments per row, fixed-order (deterministic)
  int per = rowlen >> 4;
  const float* r = buf + (size_t)t * rowlen + seg * per;
  float s = 0.0f;
  for (int j = 0; j < per; ++j) { float v = r[j]; s += v * v; }
  shpart[t * 16 + seg] = s;
  __syncthreads();
  if (tid < 32) {
    float a = 0.0f;
    for (int j = 0; j < 16; ++j) a += shpart[tid * 16 + j];
    out32[tid] = a;
  }
  __syncthreads();
}

__global__ __launch_bounds__(512) void scan_kernel(
    const half_t* __restrict__ phiK, const half_t* __restrict__ phiKT,
    const half_t* __restrict__ phiQ,
    const float* __restrict__ Vt, const float* __restrict__ kn2,
    const float* __restrict__ lr, const float* __restrict__ eta, const float* __restrict__ alpha,
    float* __restrict__ W1, float* __restrict__ W2, float* __restrict__ S1, float* __restrict__ S2,
    half_t* __restrict__ W1Tm, half_t* __restrict__ W2h, half_t* __restrict__ W2Tm,
    float* __restrict__ Z1, float* __restrict__ Hf, half_t* __restrict__ Hh,
    half_t* __restrict__ HhT,
    float* __restrict__ Df, float* __restrict__ DfT, half_t* __restrict__ Dh,
    float* __restrict__ DZf, float* __restrict__ DZfT,
    half_t* __restrict__ preds16) {
  __shared__ float shpart[512];
  __shared__ float hn2[32], dn2[32], dzn2[32];
  __shared__ float cw1[32], cs1[32], cw2[32], cs2[32];
  __shared__ float su[32], wu[32];
  __shared__ float sc[3];
  const int tid = threadIdx.x;
  const int wid = tid >> 5;
  const int lane = tid & 31;
  const int n = lane & 15, mb = (lane >> 4) << 3;

  for (int nc = 0; nc < NC; ++nc) {
    const int t0 = nc * CHUNK;

    // --- Phase A: Z1 = Kc @ W1 ; H = silu(Z1). 32 groups of 16x64 ---
    for (int g = wid; g < 32; g += 16) {
      int tm = (g >> 4) << 4, tn0 = (g & 15) << 6;
      v8f acc[4] = {};
      for (int k = 0; k < POLY; k += 32) {
        v16h a = load_a16v(phiK + (size_t)(t0 + tm) * POLY + k, POLY);
#pragma unroll
        for (int j = 0; j < 4; ++j)
          acc[j] = wmma32(a, load_b16t(W1Tm + (size_t)(tn0 + 16 * j) * POLY + k, POLY), acc[j]);
      }
#pragma unroll
      for (int j = 0; j < 4; ++j)
#pragma unroll
        for (int i = 0; i < 8; ++i) {
          int m = tm + mb + i, ng = tn0 + 16 * j + n;
          float z = acc[j][i], h = siluf_(z);
          Z1[(size_t)m * HID + ng] = z;
          Hf[(size_t)m * HID + ng] = h;
          Hh[(size_t)m * HID + ng] = (half_t)h;
          HhT[(size_t)ng * CHUNK + m] = (half_t)h;
        }
    }
    __syncthreads();
    rownorm2_512(Hf, HID, shpart, hn2);

    // --- Phase B: P = H @ W2 ; D = 2(P - V). 16 groups of 16x64 ---
    for (int g = wid; g < 16; g += 16) {
      int tm = (g >> 3) << 4, tn0 = (g & 7) << 6;
      v8f acc[4] = {};
      for (int k = 0; k < HID; k += 32) {
        v16h a = load_a16v(Hh + (size_t)tm * HID + k, HID);
#pragma unroll
        for (int j = 0; j < 4; ++j)
          acc[j] = wmma32(a, load_b16t(W2Tm + (size_t)(tn0 + 16 * j) * HID + k, HID), acc[j]);
      }
#pragma unroll
      for (int j = 0; j < 4; ++j)
#pragma unroll
        for (int i = 0; i < 8; ++i) {
          int m = tm + mb + i, ng = tn0 + 16 * j + n;
          float d = 2.0f * (acc[j][i] - Vt[(size_t)(t0 + m) * POLY + ng]);
          Df[(size_t)m * POLY + ng] = d;
          DfT[(size_t)ng * CHUNK + m] = d;
          Dh[(size_t)m * POLY + ng] = (half_t)d;
        }
    }
    __syncthreads();
    rownorm2_512(Df, POLY, shpart, dn2);

    // --- Phase C: DH = D @ W2^T ; DZ = DH * silu'(Z1). 32 groups (B = W2 row-major) ---
    for (int g = wid; g < 32; g += 16) {
      int tm = (g >> 4) << 4, tn0 = (g & 15) << 6;
      v8f acc[4] = {};
      for (int k = 0; k < POLY; k += 32) {
        v16h a = load_a16v(Dh + (size_t)tm * POLY + k, POLY);
#pragma unroll
        for (int j = 0; j < 4; ++j)
          acc[j] = wmma32(a, load_b16t(W2h + (size_t)(tn0 + 16 * j) * POLY + k, POLY), acc[j]);
      }
#pragma unroll
      for (int j = 0; j < 4; ++j)
#pragma unroll
        for (int i = 0; i < 8; ++i) {
          int m = tm + mb + i, ng = tn0 + 16 * j + n;
          float z = Z1[(size_t)m * HID + ng];
          float sg = sigmoidf_(z);
          float dz = acc[j][i] * (sg * (1.0f + z * (1.0f - sg)));
          DZf[(size_t)m * HID + ng] = dz;
          DZfT[(size_t)ng * CHUNK + m] = dz;
        }
    }
    __syncthreads();
    rownorm2_512(DZf, HID, shpart, dzn2);

    // --- Phase D: linear-recurrence coefficients (thread 0, O(32^2)) ---
    if (tid == 0) {
      float sS0 = 1.0f, wW0 = 1.0f, wS0 = 0.0f;
      for (int r = 0; r < 32; ++r) { su[r] = 0.0f; wu[r] = 0.0f; }
      for (int t = 0; t < 32; ++t) {
        float e = eta[t0 + t];
        float A = 1.0f - alpha[t0 + t];
        sS0 *= e;
        for (int r = 0; r <= t; ++r) su[r] *= e;
        su[t] += 1.0f;
        wW0 *= A; wS0 = A * wS0 + sS0;
        for (int r = 0; r <= t; ++r) wu[r] = A * wu[r] + su[r];
      }
      for (int t = 0; t < 32; ++t) {
        float l = lr[t0 + t];
        float n1 = sqrtf(kn2[t0 + t] * dzn2[t]);   // ||g1|| = ||phi_k||*||dz|| (rank-1)
        float n2 = sqrtf(hn2[t] * dn2[t]);         // ||g2|| = ||h||*||d||
        float s1 = fminf(1.0f, MAXGN / (n1 + 1e-12f));
        float s2 = fminf(1.0f, MAXGN / (n2 + 1e-12f));
        cw1[t] = -l * s1 * wu[t];  cs1[t] = -l * s1 * su[t];
        cw2[t] = -l * s2 * wu[t];  cs2[t] = -l * s2 * su[t];
      }
      sc[0] = wW0; sc[1] = wS0; sc[2] = sS0;
    }
    __syncthreads();
    const float cA = sc[0], cS0 = sc[1], cE = sc[2];

    // --- Phase E: chunk-collapsed weight updates as GEMMs ---
    // W1_new = cA*W1 + cS0*S1 + PhiK^T diag(cw1) DZ ; S1_new = cE*S1 + PhiK^T diag(cs1) DZ
    // 512 groups of 16x64 over the 512x1024 W1
    for (int g = wid; g < 512; g += 16) {
      int tm = (g >> 4) << 4, tn0 = (g & 15) << 6;
      v16h aT = load_a16v(phiKT + (size_t)tm * TT + t0, TT);
#pragma unroll
      for (int j = 0; j < 4; ++j) {
        int n0 = tn0 + 16 * j;
        v8f accW = wmma32(aT, load_b16sT(DZfT, n0, cw1), v8f{});
        v8f accS = wmma32(aT, load_b16sT(DZfT, n0, cs1), v8f{});
#pragma unroll
        for (int i = 0; i < 8; ++i) {
          int mm = tm + mb + i, nn = n0 + n;
          size_t idx = (size_t)mm * HID + nn;
          float wo = W1[idx], so = S1[idx];
          float sn = cE * so + accS[i];
          float wn = cA * wo + cS0 * so + accW[i];
          S1[idx] = sn; W1[idx] = wn;
          W1Tm[(size_t)nn * POLY + mm] = (half_t)wn;
        }
      }
    }
    // W2_new = cA*W2 + cS0*S2 + H^T diag(cw2) D ; 512 groups over 1024x512 W2
    for (int g = wid; g < 512; g += 16) {
      int tm = (g >> 3) << 4, tn0 = (g & 7) << 6;
      v16h aT = load_a16v(HhT + (size_t)tm * CHUNK, CHUNK);
#pragma unroll
      for (int j = 0; j < 4; ++j) {
        int n0 = tn0 + 16 * j;
        v8f accW = wmma32(aT, load_b16sT(DfT, n0, cw2), v8f{});
        v8f accS = wmma32(aT, load_b16sT(DfT, n0, cs2), v8f{});
#pragma unroll
        for (int i = 0; i < 8; ++i) {
          int mm = tm + mb + i, nn = n0 + n;
          size_t idx = (size_t)mm * POLY + nn;
          float wo = W2[idx], so = S2[idx];
          float sn = cE * so + accS[i];
          float wn = cA * wo + cS0 * so + accW[i];
          S2[idx] = sn; W2[idx] = wn;
          W2h[idx] = (half_t)wn;
          W2Tm[(size_t)nn * HID + mm] = (half_t)wn;
        }
      }
    }
    __syncthreads();

    // --- Phase F: Hq = silu(Qc @ W1_new) (reuses Hh) ---
    for (int g = wid; g < 32; g += 16) {
      int tm = (g >> 4) << 4, tn0 = (g & 15) << 6;
      v8f acc[4] = {};
      for (int k = 0; k < POLY; k += 32) {
        v16h a = load_a16v(phiQ + (size_t)(t0 + tm) * POLY + k, POLY);
#pragma unroll
        for (int j = 0; j < 4; ++j)
          acc[j] = wmma32(a, load_b16t(W1Tm + (size_t)(tn0 + 16 * j) * POLY + k, POLY), acc[j]);
      }
#pragma unroll
      for (int j = 0; j < 4; ++j)
#pragma unroll
        for (int i = 0; i < 8; ++i)
          Hh[(size_t)(tm + mb + i) * HID + tn0 + 16 * j + n] = (half_t)siluf_(acc[j][i]);
    }
    __syncthreads();

    // --- Phase G: preds = Hq @ W2_new ---
    for (int g = wid; g < 16; g += 16) {
      int tm = (g >> 3) << 4, tn0 = (g & 7) << 6;
      v8f acc[4] = {};
      for (int k = 0; k < HID; k += 32) {
        v16h a = load_a16v(Hh + (size_t)tm * HID + k, HID);
#pragma unroll
        for (int j = 0; j < 4; ++j)
          acc[j] = wmma32(a, load_b16t(W2Tm + (size_t)(tn0 + 16 * j) * HID + k, HID), acc[j]);
      }
#pragma unroll
      for (int j = 0; j < 4; ++j)
#pragma unroll
        for (int i = 0; i < 8; ++i)
          preds16[(size_t)(t0 + tm + mb + i) * POLY + tn0 + 16 * j + n] = (half_t)acc[j][i];
    }
    __syncthreads();
  }
}

// ---------------- host launch ----------------

extern "C" void kernel_launch(void* const* d_in, const int* in_sizes, int n_in,
                              void* d_out, int out_size, void* d_ws, size_t ws_size,
                              hipStream_t stream) {
  const float* x      = (const float*)d_in[0];
  const float* pers   = (const float*)d_in[1];
  const float* Wk     = (const float*)d_in[2];
  const float* Wq     = (const float*)d_in[3];
  const float* Wv     = (const float*)d_in[4];
  const float* Wo     = (const float*)d_in[5];
  const float* ckw    = (const float*)d_in[6];
  const float* cqw    = (const float*)d_in[7];
  const float* cvw    = (const float*)d_in[8];
  const float* ckb    = (const float*)d_in[9];
  const float* cqb    = (const float*)d_in[10];
  const float* cvb    = (const float*)d_in[11];
  const float* lrw    = (const float*)d_in[12];
  const float* lrb    = (const float*)d_in[13];
  const float* momw   = (const float*)d_in[14];
  const float* momb   = (const float*)d_in[15];
  const float* decw   = (const float*)d_in[16];
  const float* decb   = (const float*)d_in[17];
  const float* gw     = (const float*)d_in[18];
  const float* gb     = (const float*)d_in[19];
  const float* polyc  = (const float*)d_in[20];
  const float* vexp   = (const float*)d_in[21];
  const float* memout = (const float*)d_in[22];
  const float* MW1    = (const float*)d_in[23];
  const float* MW2    = (const float*)d_in[24];
  (void)in_sizes; (void)n_in; (void)out_size; (void)ws_size;

  char* base = (char*)d_ws;
  size_t cur = 0;
  auto carve = [&](size_t bytes) -> void* {
    void* r = base + cur;
    cur += (bytes + 255) & ~(size_t)255;
    return r;
  };

  float*  xf32    = (float*) carve((size_t)TT * DIM * 4);
  half_t* xf16    = (half_t*)carve((size_t)TT * DIM * 2);
  half_t* WkT16   = (half_t*)carve((size_t)DIM * MEM * 2);
  half_t* WqT16   = (half_t*)carve((size_t)DIM * MEM * 2);
  half_t* WvT16   = (half_t*)carve((size_t)DIM * MEM * 2);
  half_t* WoT16   = (half_t*)carve((size_t)MEM * DIM * 2);
  half_t* vexpT16 = (half_t*)carve((size_t)MEM * POLY * 2);
  half_t* moT16   = (half_t*)carve((size_t)POLY * MEM * 2);
  float*  kpre    = (float*) carve((size_t)TT * MEM * 4);
  float*  qpre    = (float*) carve((size_t)TT * MEM * 4);
  float*  vpre    = (float*) carve((size_t)TT * MEM * 4);
  half_t* phiK    = (half_t*)carve((size_t)TT * POLY * 2);
  half_t* phiKT   = (half_t*)carve((size_t)TT * POLY * 2);
  half_t* phiQ    = (half_t*)carve((size_t)TT * POLY * 2);
  half_t* v16b    = (half_t*)carve((size_t)TT * MEM * 2);
  float*  Vt      = (float*) carve((size_t)TT * POLY * 4);
  float*  kn2     = (float*) carve((size_t)TT * 4);
  float*  lr      = (float*) carve((size_t)TT * 4);
  float*  eta     = (float*) carve((size_t)TT * 4);
  float*  alpha   = (float*) carve((size_t)TT * 4);
  float*  gate    = (float*) carve((size_t)TT * 4);
  float*  W1      = (float*) carve((size_t)POLY * HID * 4);
  float*  W2      = (float*) carve((size_t)HID * POLY * 4);
  float*  S1      = (float*) carve((size_t)POLY * HID * 4);
  float*  S2      = (float*) carve((size_t)HID * POLY * 4);
  half_t* W1Tm    = (half_t*)carve((size_t)POLY * HID * 2);
  half_t* W2h     = (half_t*)carve((size_t)HID * POLY * 2);
  half_t* W2Tm    = (half_t*)carve((size_t)POLY * HID * 2);
  float*  Z1      = (float*) carve((size_t)CHUNK * HID * 4);
  float*  Hf      = (float*) carve((size_t)CHUNK * HID * 4);
  half_t* Hh      = (half_t*)carve((size_t)CHUNK * HID * 2);
  half_t* HhT     = (half_t*)carve((size_t)HID * CHUNK * 2);
  float*  Df      = (float*) carve((size_t)CHUNK * POLY * 4);
  float*  DfT     = (float*) carve((size_t)POLY * CHUNK * 4);
  half_t* Dh      = (half_t*)carve((size_t)CHUNK * POLY * 2);
  float*  DZf     = (float*) carve((size_t)CHUNK * HID * 4);
  float*  DZfT    = (float*) carve((size_t)HID * CHUNK * 4);
  half_t* preds16 = (half_t*)carve((size_t)TT * POLY * 2);
  float*  retr    = (float*) carve((size_t)TT * MEM * 4);
  half_t* retr16  = (half_t*)carve((size_t)TT * MEM * 2);
  float*  ybuf    = (float*) carve((size_t)TT * DIM * 4);

  auto gemm = [&](const half_t* A, const half_t* BT, float* C, int M, int N, int K) {
    int waves = (M / 16) * (N / 64);
    gemm_f16_kernel<<<waves / 8, 256, 0, stream>>>(A, BT, C, M, N, K);
  };

  // setup
  build_xf_kernel<<<(TT * DIM + 255) / 256, 256, 0, stream>>>(x, pers, xf32, xf16);
  f32_to_f16_T_kernel<<<(DIM * MEM + 255) / 256, 256, 0, stream>>>(Wk, WkT16, DIM, MEM);
  f32_to_f16_T_kernel<<<(DIM * MEM + 255) / 256, 256, 0, stream>>>(Wq, WqT16, DIM, MEM);
  f32_to_f16_T_kernel<<<(DIM * MEM + 255) / 256, 256, 0, stream>>>(Wv, WvT16, DIM, MEM);
  f32_to_f16_T_kernel<<<(MEM * DIM + 255) / 256, 256, 0, stream>>>(Wo, WoT16, MEM, DIM);
  f32_to_f16_T_kernel<<<(MEM * POLY + 255) / 256, 256, 0, stream>>>(vexp, vexpT16, MEM, POLY);
  f32_to_f16_T_kernel<<<(POLY * MEM + 255) / 256, 256, 0, stream>>>(memout, moT16, POLY, MEM);
  init_mem_kernel<<<(POLY * HID + 255) / 256, 256, 0, stream>>>(MW1, MW2, W1, W2, S1, S2,
                                                                W1Tm, W2h, W2Tm);
  // projections (WMMA, b128 fragment loads)
  gemm(xf16, WkT16, kpre, TT, MEM, DIM);
  gemm(xf16, WqT16, qpre, TT, MEM, DIM);
  gemm(xf16, WvT16, vpre, TT, MEM, DIM);
  // causal depthwise conv + silu + polynomial features (+ transposed phiK copy)
  conv_poly_kernel<<<(TT * MEM + 255) / 256, 256, 0, stream>>>(
      kpre, qpre, vpre, ckw, cqw, cvw, ckb, cqb, cvb, polyc, phiK, phiKT, phiQ, v16b);
  kn2_kernel<<<TT, 256, 0, stream>>>(phiK, kn2);
  gates_kernel<<<TT, 256, 0, stream>>>(xf32, lrw, lrb, momw, momb, decw, decb, gw, gb,
                                       lr, eta, alpha, gate);
  gemm(v16b, vexpT16, Vt, TT, POLY, MEM);   // v target expansion (WMMA)
  // sequential chunk scan (persistent single workgroup, 16 waves, WMMA throughout)
  scan_kernel<<<1, 512, 0, stream>>>(phiK, phiKT, phiQ, Vt, kn2, lr, eta, alpha,
                                     W1, W2, S1, S2, W1Tm, W2h, W2Tm,
                                     Z1, Hf, Hh, HhT, Df, DfT, Dh, DZf, DZfT, preds16);
  // output projections (WMMA)
  gemm(preds16, moT16, retr, TT, MEM, POLY);
  gatefold_kernel<<<(TT * MEM + 255) / 256, 256, 0, stream>>>(retr, gate, retr16);
  gemm(retr16, WoT16, ybuf, TT, DIM, MEM);
  copyout_kernel<<<(SEQ * DIM + 255) / 256, 256, 0, stream>>>(ybuf, (float*)d_out);
}
